// GEN_48962627174857
// MI455X (gfx1250) — compile-verified, matrix operand order checked by reading, not run on hardware
//
#include <hip/hip_runtime.h>
#include <hip/hip_bf16.h>

// ---------------------------------------------------------------------------
// Types for CDNA5 WMMA (wave32): A/B = 16 bf16 per lane, C/D = 8 f32 per lane
// ---------------------------------------------------------------------------
typedef __attribute__((ext_vector_type(16))) __bf16 v16bf;
typedef __attribute__((ext_vector_type(8)))  float  v8f;
typedef unsigned int u32;
typedef __attribute__((ext_vector_type(4))) u32 u32x4;

union FragU { v16bf bf; u32x4 q[2]; };

__device__ inline __bf16 f2bf(float f) {
  u32 u = __builtin_bit_cast(u32, f);
  u32 r = u + 0x7FFFu + ((u >> 16) & 1u);   // round-to-nearest-even
  unsigned short s = (unsigned short)(r >> 16);
  return __builtin_bit_cast(__bf16, s);
}

// CDNA5 16-bit A fragment (16x32 MxK), lanes 0-15: M=lane, elems 0-7 = K 0..7,
// elems 8-15 = K 16..23; lanes 16-31 shift K by +8. The B fragment (32x16 KxN)
// uses the same pattern indexed by N, so loading from a row-major [N][K]
// (i.e. pre-transposed B) is identical code. Works for global or LDS pointers.
__device__ inline v16bf frag_rowmajor(const __bf16* p, int stride) {
  const int l = threadIdx.x & 31, r = l & 15, hl = l >> 4;
  FragU f;
  const __bf16* q = p + (size_t)r * stride + hl * 8;
  f.q[0] = *(const u32x4*)(q);
  f.q[1] = *(const u32x4*)(q + 16);
  return f.bf;
}

__device__ inline v8f wmma_bf16(v16bf a, v16bf b, v8f c) {
  // D = A(16x32 bf16) x B(32x16 bf16) + C(16x16 f32)
  return __builtin_amdgcn_wmma_f32_16x16x32_bf16(
      false, a, false, b, (short)0, c, false, false);
}

// ---------------------------------------------------------------------------
// LayerNorm: one block per 1024-wide row, fp32 in -> bf16 out
// ---------------------------------------------------------------------------
__global__ __launch_bounds__(256) void ln_bf16_k(
    const float* __restrict__ x, const float* __restrict__ w,
    const float* __restrict__ b, __bf16* __restrict__ out)
{
  const int row = blockIdx.x, t = threadIdx.x;
  const float* xr = x + (size_t)row * 1024;
  float v0 = xr[t], v1 = xr[t + 256], v2 = xr[t + 512], v3 = xr[t + 768];
  float s  = v0 + v1 + v2 + v3;
  float ss = v0 * v0 + v1 * v1 + v2 * v2 + v3 * v3;
  for (int m = 1; m < 32; m <<= 1) {
    s  += __shfl_xor(s,  m, 32);
    ss += __shfl_xor(ss, m, 32);
  }
  __shared__ float rs[8], rss[8];
  __shared__ float mu_s, inv_s;
  if ((t & 31) == 0) { rs[t >> 5] = s; rss[t >> 5] = ss; }
  __syncthreads();
  if (t == 0) {
    float S = 0.f, SS = 0.f;
    for (int i = 0; i < 8; i++) { S += rs[i]; SS += rss[i]; }
    float mu = S * (1.0f / 1024.0f);
    float var = SS * (1.0f / 1024.0f) - mu * mu;
    mu_s = mu; inv_s = rsqrtf(var + 1e-5f);
  }
  __syncthreads();
  const float mu = mu_s, inv = inv_s;
  __bf16* o = out + (size_t)row * 1024;
  o[t      ] = f2bf((v0 - mu) * inv * w[t      ] + b[t      ]);
  o[t + 256] = f2bf((v1 - mu) * inv * w[t + 256] + b[t + 256]);
  o[t + 512] = f2bf((v2 - mu) * inv * w[t + 512] + b[t + 512]);
  o[t + 768] = f2bf((v3 - mu) * inv * w[t + 768] + b[t + 768]);
}

// fp32 -> bf16 straight convert
__global__ __launch_bounds__(256) void conv_bf16_k(
    const float* __restrict__ in, __bf16* __restrict__ out, int n)
{
  int i = blockIdx.x * 256 + threadIdx.x;
  if (i < n) out[i] = f2bf(in[i]);
}

// fp32 W[K][N] -> bf16 Wt[N][K]  (B-operand pre-transpose)
__global__ __launch_bounds__(256) void transp_bf16_k(
    const float* __restrict__ in, __bf16* __restrict__ out, int K, int N)
{
  size_t i = (size_t)blockIdx.x * 256 + threadIdx.x;
  if (i < (size_t)K * N) {
    int n = (int)(i / K), k = (int)(i % K);
    out[i] = f2bf(in[(size_t)k * N + n]);
  }
}

// ---------------------------------------------------------------------------
// Tiled WMMA GEMM: C[MxN] = A[MxK] (bf16 row-major) @ Bt[NxK] (bf16, B^T)
// 128 threads = 4 waves; block tile 128x128, each wave a 64x64 sub-tile
// (4x4 WMMA accumulators -> 16 WMMAs per 32-wide K step, 1 LDS frag read
// per 2 WMMAs). Global loads for the next K step are register-staged while the
// current step's WMMAs run (software pipeline).
// MODE 0: bf16 out    MODE 1: gelu(exact)->bf16    MODE 2: *tanh(gate)+R ->f32
// MODE 3: +R -> f32   MODE 4: KV split (K natural bf16, V transposed bf16)
// ---------------------------------------------------------------------------
#define TM 128
#define TN 128
#define TK 32

template <int MODE>
__global__ __launch_bounds__(128) void gemm_bf16_k(
    const __bf16* __restrict__ A, const __bf16* __restrict__ Bt,
    int M, int N, int K,
    __bf16* __restrict__ Obf, float* __restrict__ Of,
    const float* __restrict__ R, const float* __restrict__ gate,
    __bf16* __restrict__ Kout, __bf16* __restrict__ VTout)
{
  __shared__ __bf16 sA[TM * TK];   // 8 KB
  __shared__ __bf16 sB[TN * TK];   // 8 KB
  const int t = threadIdx.x;
  const int wave = t >> 5;
  const int m0 = blockIdx.y * TM;
  const int n0 = blockIdx.x * TN;
  const int wm = (wave & 1) * 64;
  const int wn = (wave >> 1) * 64;

  v8f acc[4][4] = {};
  u32x4 ra[4], rb[4];

  // each 128x32 bf16 tile = 512 16B chunks; 128 threads x 4 chunks
  const int rowc = t >> 2;              // base row for chunk 0 (t + i*128)
  const int cc   = (t & 3) * 8;         // k-subchunk (8 bf16 = 16B)

#define LOAD_TILE(K0)                                                         \
  {                                                                           \
    _Pragma("unroll") for (int i = 0; i < 4; i++) {                           \
      int row = rowc + i * 32;                                                \
      ra[i] = *(const u32x4*)(A + (size_t)(m0 + row) * K + (K0) + cc);        \
      rb[i] = *(const u32x4*)(Bt + (size_t)(n0 + row) * K + (K0) + cc);       \
    }                                                                         \
  }
#define STORE_TILE()                                                          \
  {                                                                           \
    _Pragma("unroll") for (int i = 0; i < 4; i++) {                           \
      int row = rowc + i * 32;                                                \
      *(u32x4*)(sA + row * TK + cc) = ra[i];                                  \
      *(u32x4*)(sB + row * TK + cc) = rb[i];                                  \
    }                                                                         \
  }

  LOAD_TILE(0)
  STORE_TILE()

  for (int k0 = 0; k0 < K; k0 += TK) {
    __syncthreads();                       // LDS tile for k0 is ready
    const bool more = (k0 + TK) < K;
    if (more) LOAD_TILE(k0 + TK)           // prefetch next tile into regs

    v16bf bfr[4];
#pragma unroll
    for (int j = 0; j < 4; j++)
      bfr[j] = frag_rowmajor(sB + (wn + j * 16) * TK, TK);
#pragma unroll
    for (int i = 0; i < 4; i++) {
      v16bf af = frag_rowmajor(sA + (wm + i * 16) * TK, TK);
#pragma unroll
      for (int j = 0; j < 4; j++)
        acc[i][j] = wmma_bf16(af, bfr[j], acc[i][j]);
    }
    __syncthreads();                       // everyone done reading LDS
    if (more) STORE_TILE()
  }
#undef LOAD_TILE
#undef STORE_TILE

  const int l = t & 31, lr = l & 15, hl = l >> 4;
  float g = 0.f;
  if (MODE == 2) g = tanhf(gate[0]);

#pragma unroll
  for (int i = 0; i < 4; i++) {
#pragma unroll
    for (int j = 0; j < 4; j++) {
#pragma unroll
      for (int r = 0; r < 8; r++) {
        int row = m0 + wm + i * 16 + r + 8 * hl;
        int col = n0 + wn + j * 16 + lr;
        float v = acc[i][j][r];
        if (MODE == 0) {
          Obf[(size_t)row * N + col] = f2bf(v);
        } else if (MODE == 1) {
          float ge = 0.5f * v * (1.0f + erff(v * 0.70710678118f));
          Obf[(size_t)row * N + col] = f2bf(ge);
        } else if (MODE == 2) {
          Of[(size_t)row * N + col] = v * g + R[(size_t)row * N + col];
        } else if (MODE == 3) {
          Of[(size_t)row * N + col] = v + R[(size_t)row * N + col];
        } else {
          if (col < 1024) {
            Kout[(size_t)row * 1024 + col] = f2bf(v);
          } else {
            int c2 = col - 1024, hh = c2 >> 6, dd = c2 & 63;
            int bb = row >> 10, mr = row & 1023;
            VTout[(((size_t)(bb * 16 + hh)) * 64 + dd) * 1024 + mr] = f2bf(v);
          }
        }
      }
    }
  }
}

// ---------------------------------------------------------------------------
// Flash-style cross attention. Grid: (n/64 tiles, b*h). 4 waves; each wave
// owns 16 query rows. K chunked 32 keys at a time; online softmax with
// shfl_xor reductions over the 16-lane halves (C-layout rows M = r + 8*hl);
// P converted to A-fragment layout via per-wave LDS round trip.
// Q,K: bf16 (b*n, 1024) / (b*m, 1024) with head at col h*64. VT: (b,h,64,1024).
// ---------------------------------------------------------------------------
__global__ __launch_bounds__(128) void attn_k(
    const __bf16* __restrict__ Q, const __bf16* __restrict__ Kb,
    const __bf16* __restrict__ VT, __bf16* __restrict__ O)
{
  __shared__ __bf16 sP[4 * 16 * 32];
  const int t = threadIdx.x, wave = t >> 5, l = t & 31, lr = l & 15, hl = l >> 4;
  const int bh = blockIdx.y, b = bh >> 4, head = bh & 15;
  const int qbase = blockIdx.x * 64 + wave * 16;
  __bf16* lP = sP + wave * (16 * 32);
  const float scale = 0.125f;   // 64^-0.5

  const __bf16* qp = Q + ((size_t)(b * 2048 + qbase)) * 1024 + head * 64;
  v16bf q0 = frag_rowmajor(qp, 1024);        // K (=d) 0..31
  v16bf q1 = frag_rowmajor(qp + 32, 1024);   // K 32..63

  v8f o0 = {}, o1 = {}, o2 = {}, o3 = {};
  float Mr[8], Sr[8];
#pragma unroll
  for (int r = 0; r < 8; r++) { Mr[r] = -1e30f; Sr[r] = 0.f; }

  for (int mc = 0; mc < 1024; mc += 32) {
    const __bf16* kp = Kb + ((size_t)(b * 1024 + mc)) * 1024 + head * 64;
    v8f s0 = {}, s1 = {};
    s0 = wmma_bf16(q0, frag_rowmajor(kp, 1024), s0);
    s0 = wmma_bf16(q1, frag_rowmajor(kp + 32, 1024), s0);
    s1 = wmma_bf16(q0, frag_rowmajor(kp + 16 * 1024, 1024), s1);
    s1 = wmma_bf16(q1, frag_rowmajor(kp + 16 * 1024 + 32, 1024), s1);

#pragma unroll
    for (int r = 0; r < 8; r++) {
      float a = s0[r] * scale, c = s1[r] * scale;
      float mx = fmaxf(a, c);
      mx = fmaxf(mx, __shfl_xor(mx, 1, 32));
      mx = fmaxf(mx, __shfl_xor(mx, 2, 32));
      mx = fmaxf(mx, __shfl_xor(mx, 4, 32));
      mx = fmaxf(mx, __shfl_xor(mx, 8, 32));
      float Mn = fmaxf(Mr[r], mx);
      float corr = __expf(Mr[r] - Mn);
      float p0 = __expf(a - Mn), p1 = __expf(c - Mn);
      float rsum = p0 + p1;
      rsum += __shfl_xor(rsum, 1, 32);
      rsum += __shfl_xor(rsum, 2, 32);
      rsum += __shfl_xor(rsum, 4, 32);
      rsum += __shfl_xor(rsum, 8, 32);
      Sr[r] = Sr[r] * corr + rsum;
      Mr[r] = Mn;
      o0[r] *= corr; o1[r] *= corr; o2[r] *= corr; o3[r] *= corr;
      lP[(r + 8 * hl) * 32 + lr]      = f2bf(p0);
      lP[(r + 8 * hl) * 32 + 16 + lr] = f2bf(p1);
    }
    __syncthreads();
    v16bf pf = frag_rowmajor(lP, 32);
    const __bf16* vp = VT + ((size_t)bh * 64) * 1024 + mc;
    o0 = wmma_bf16(pf, frag_rowmajor(vp            , 1024), o0);
    o1 = wmma_bf16(pf, frag_rowmajor(vp + 16 * 1024, 1024), o1);
    o2 = wmma_bf16(pf, frag_rowmajor(vp + 32 * 1024, 1024), o2);
    o3 = wmma_bf16(pf, frag_rowmajor(vp + 48 * 1024, 1024), o3);
    __syncthreads();
  }

#pragma unroll
  for (int r = 0; r < 8; r++) {
    float inv = 1.0f / Sr[r];
    size_t row = (size_t)(b * 2048 + qbase + r + 8 * hl);
    __bf16* op = O + row * 1024 + head * 64;
    op[ 0 + lr] = f2bf(o0[r] * inv);
    op[16 + lr] = f2bf(o1[r] * inv);
    op[32 + lr] = f2bf(o2[r] * inv);
    op[48 + lr] = f2bf(o3[r] * inv);
  }
}

// ---------------------------------------------------------------------------
extern "C" void kernel_launch(void* const* d_in, const int* in_sizes, int n_in,
                              void* d_out, int out_size, void* d_ws, size_t ws_size,
                              hipStream_t stream)
{
  (void)in_sizes; (void)n_in; (void)out_size; (void)ws_size;
  const float* x     = (const float*)d_in[0];
  const float* media = (const float*)d_in[1];
  const float* ln_w  = (const float*)d_in[2];
  const float* ln_b  = (const float*)d_in[3];
  const float* Wq    = (const float*)d_in[4];
  const float* Wkv   = (const float*)d_in[5];
  const float* Wo    = (const float*)d_in[6];
  const float* gate  = (const float*)d_in[7];
  const float* ffw   = (const float*)d_in[8];
  const float* ffb   = (const float*)d_in[9];
  const float* W1    = (const float*)d_in[10];
  const float* W2    = (const float*)d_in[11];
  float* out = (float*)d_out;

  const int BN = 8192;   // b*n tokens
  const int BM = 4096;   // b*m media tokens
  const int D  = 1024;
  const int FF = 4096;

  char* base = (char*)d_ws;
  size_t off = 0;
  auto alloc = [&](size_t bytes) -> void* {
    void* p = base + off;
    off += (bytes + 255) & ~(size_t)255;
    return p;
  };

  __bf16* xn    = (__bf16*)alloc((size_t)BN * D * 2);
  __bf16* medb  = (__bf16*)alloc((size_t)BM * D * 2);
  __bf16* WqT   = (__bf16*)alloc((size_t)D * D * 2);
  __bf16* WkvT  = (__bf16*)alloc((size_t)D * 2 * D * 2);
  __bf16* WoT   = (__bf16*)alloc((size_t)D * D * 2);
  __bf16* W1T   = (__bf16*)alloc((size_t)D * FF * 2);
  __bf16* W2T   = (__bf16*)alloc((size_t)FF * D * 2);
  __bf16* qb    = (__bf16*)alloc((size_t)BN * D * 2);
  __bf16* kb    = (__bf16*)alloc((size_t)BM * D * 2);
  __bf16* vT    = (__bf16*)alloc((size_t)BM * D * 2);
  __bf16* ao    = (__bf16*)alloc((size_t)BN * D * 2);
  float*  x2    = (float*)alloc((size_t)BN * D * 4);
  __bf16* hin   = (__bf16*)alloc((size_t)BN * D * 2);
  __bf16* h1    = (__bf16*)alloc((size_t)BN * FF * 2);

  // 1) LN(x) -> bf16, media -> bf16, weights -> transposed bf16
  ln_bf16_k<<<BN, 256, 0, stream>>>(x, ln_w, ln_b, xn);
  conv_bf16_k<<<(BM * D) / 256, 256, 0, stream>>>(media, medb, BM * D);
  transp_bf16_k<<<(D * D) / 256, 256, 0, stream>>>(Wq, WqT, D, D);
  transp_bf16_k<<<(D * 2 * D) / 256, 256, 0, stream>>>(Wkv, WkvT, D, 2 * D);
  transp_bf16_k<<<(D * D) / 256, 256, 0, stream>>>(Wo, WoT, D, D);
  transp_bf16_k<<<(D * FF) / 256, 256, 0, stream>>>(W1, W1T, D, FF);
  transp_bf16_k<<<(FF * D) / 256, 256, 0, stream>>>(W2, W2T, FF, D);

  // 2) Q = xn @ Wq ; KV = media @ Wkv (K natural, V transposed)
  gemm_bf16_k<0><<<dim3(D / TN, BN / TM), 128, 0, stream>>>(
      xn, WqT, BN, D, D, qb, nullptr, nullptr, nullptr, nullptr, nullptr);
  gemm_bf16_k<4><<<dim3((2 * D) / TN, BM / TM), 128, 0, stream>>>(
      medb, WkvT, BM, 2 * D, D, nullptr, nullptr, nullptr, nullptr, kb, vT);

  // 3) attention
  attn_k<<<dim3(2048 / 64, 64), 128, 0, stream>>>(qb, kb, vT, ao);

  // 4) x2 = (attn_out @ Wo) * tanh(gate) + x
  gemm_bf16_k<2><<<dim3(D / TN, BN / TM), 128, 0, stream>>>(
      ao, WoT, BN, D, D, nullptr, x2, x, gate, nullptr, nullptr);

  // 5) FFN: LN -> W1 -> gelu -> W2 -> + x2
  ln_bf16_k<<<BN, 256, 0, stream>>>(x2, ffw, ffb, hin);
  gemm_bf16_k<1><<<dim3(FF / TN, BN / TM), 128, 0, stream>>>(
      hin, W1T, BN, FF, D, h1, nullptr, nullptr, nullptr, nullptr, nullptr);
  gemm_bf16_k<3><<<dim3(D / TN, BN / TM), 128, 0, stream>>>(
      h1, W2T, BN, D, FF, nullptr, out, x2, nullptr, nullptr, nullptr);
}
